// CrossCorrelationFFT_36120674959508
// MI455X (gfx1250) — compile-verified, matrix operand order checked by reading, not run on hardware
//
#include <hip/hip_runtime.h>
#include <hip/hip_bf16.h>
#include <stddef.h>
#include <stdint.h>

typedef _Float16 half_t;
typedef __attribute__((ext_vector_type(16))) _Float16 v16h;
typedef __attribute__((ext_vector_type(8)))  _Float16 h8;
typedef __attribute__((ext_vector_type(4)))  _Float16 h4;
typedef __attribute__((ext_vector_type(8)))  float    v8f;
typedef __attribute__((ext_vector_type(4)))  unsigned int u32x4;
typedef __attribute__((ext_vector_type(8)))  int      i32x8;
typedef __attribute__((ext_vector_type(4)))  int      i32x4;

#define BB 4
#define CC 32
#define HH 256
#define WW 256
#define MS 10
#define NDX 21
#define NPAIR 528
#define NCPY 4          // pre-shifted copies, r = shift mod 4
#define RS 288          // global row stride with circular halo
#define BSTR 264        // LDS B-row stride in halves (512B row + 16B TDM pad)

// ---------------------------------------------------------------------------
// Kernel 1: per-(b,c) mean and scale = 1/(std*sqrt(N))  (std<eps -> 0)
// ---------------------------------------------------------------------------
__global__ void k_stats(const float* __restrict__ x, float* __restrict__ stats) {
    int b = blockIdx.x;
    int t = threadIdx.x;
    int c = t & 31;
    int sub = t >> 5;
    float s = 0.f, s2 = 0.f;
    for (int h = 0; h < HH; ++h) {
        for (int w = sub; w < WW; w += 8) {
            float v = x[(((size_t)b * HH + h) * WW + w) * CC + c];
            s += v; s2 += v * v;
        }
    }
    __shared__ float red[2][8][32];
    red[0][sub][c] = s;
    red[1][sub][c] = s2;
    __syncthreads();
    if (sub == 0) {
        float S = 0.f, S2 = 0.f;
        for (int k = 0; k < 8; ++k) { S += red[0][k][c]; S2 += red[1][k][c]; }
        const float N = (float)(HH * WW);
        float mean = S / N;
        float var  = fmaxf(S2 / N - mean * mean, 0.f);
        float sd   = sqrtf(var);
        float sc   = (sd < 1e-9f) ? 0.f : 1.f / (sd * 256.f);
        stats[b * 32 + c]       = mean;
        stats[128 + b * 32 + c] = sc;
    }
}

// ---------------------------------------------------------------------------
// Kernel 2: materialize 4 x-shifted f16 copies with circular halo rows:
//   cpy[r][b][c][h][wp] = xs[b][c][h][(wp - 16 + r) & 255],  wp in [0,288)
// ---------------------------------------------------------------------------
__global__ void k_copies(const float* __restrict__ x, const float* __restrict__ stats,
                         half_t* __restrict__ cpy) {
    int h = blockIdx.x;
    int b = blockIdx.y;
    __shared__ float row[32][257];
    int t = threadIdx.x;
    for (int i = 0; i < 32; ++i) {
        int idx = t + 256 * i;
        int c = idx & 31, w = idx >> 5;
        row[c][w] = x[(((size_t)b * HH + h) * WW + w) * CC + c];
    }
    __syncthreads();
    int c   = t >> 3;
    int ws0 = (t & 7) * 36;
    float mean = stats[b * 32 + c];
    float sc   = stats[128 + b * 32 + c];
    for (int r = 0; r < NCPY; ++r) {
        size_t base = ((((size_t)r * BB + b) * CC + c) * HH + h) * RS;
        for (int k = 0; k < 36; ++k) {
            int wp   = ws0 + k;
            int wsrc = (wp - 16 + r) & 255;
            cpy[base + wp] = (half_t)((row[c][wsrc] - mean) * sc);
        }
    }
}

// ---------------------------------------------------------------------------
// Kernel 3: zero output
// ---------------------------------------------------------------------------
__global__ void k_zero(float* __restrict__ p, int n) {
    int i = blockIdx.x * 256 + threadIdx.x;
    if (i < n) p[i] = 0.f;
}

// ---------------------------------------------------------------------------
// Fragment assembly helpers
// ---------------------------------------------------------------------------
__device__ __forceinline__ v16h combine8(h8 lo, h8 hi) {
    return __builtin_shufflevector(lo, hi, 0,1,2,3,4,5,6,7,8,9,10,11,12,13,14,15);
}
__device__ __forceinline__ h8 combine4(h4 a, h4 b) {
    return __builtin_shufflevector(a, b, 0,1,2,3,4,5,6,7);
}
__device__ __forceinline__ v16h loadA(const half_t* a, int kb) {
    return combine8(combine4(*(const h4*)(a + kb),      *(const h4*)(a + kb + 4)),
                    combine4(*(const h4*)(a + kb + 16), *(const h4*)(a + kb + 20)));
}

// ---------------------------------------------------------------------------
// TDM: DMA one 32ch x 256half B-row tile (global, stride HH*RS halves) into
// LDS with 16B padding after each 512B row -> LDS stride 264 halves.
// ---------------------------------------------------------------------------
__device__ __forceinline__ void tdm_stage(const half_t* gsrc, unsigned int lds_byte) {
    uint64_t ga = (uint64_t)(uintptr_t)gsrc;
    u32x4 g0;
    g0[0] = 1u;                                   // count=1, user descriptor
    g0[1] = lds_byte;                             // lds_addr
    g0[2] = (unsigned int)ga;                     // global_addr[31:0]
    g0[3] = (unsigned int)(ga >> 32) | (2u << 30);// global_addr[56:32] | type=2
    i32x8 g1;
    const unsigned int TD = 1u << 20;             // generous tensor dims (no OOB clip)
    g1[0] = (1 << 16)        // data_size = 2 bytes
          | (1 << 20)        // pad_enable
          | (6 << 22)        // pad_interval: 128 DWORDs (512B) between pads
          | (3 << 25);       // pad_amount: 4 DWORDs (16B)
    g1[1] = (int)((TD & 0xFFFFu) << 16);          // tensor_dim0[15:0]
    g1[2] = (int)((TD >> 16) | ((TD & 0xFFFFu) << 16));   // td0[31:16] | td1[15:0]
    g1[3] = (int)((TD >> 16) | (256u << 16));     // td1[31:16] | tile_dim0=256
    g1[4] = 32;                                   // tile_dim1=32, tile_dim2=0
    g1[5] = HH * RS;                              // tensor_dim0_stride (halves)
    g1[6] = 0;
    g1[7] = 0;
    i32x4 gz4 = {0, 0, 0, 0};                     // 2-D tensor: groups 2/3 unused
    i32x8 gz8 = {0, 0, 0, 0, 0, 0, 0, 0};
    __builtin_amdgcn_tensor_load_to_lds(g0, g1, gz4, gz4, gz8, 0);
}

// ---------------------------------------------------------------------------
// Kernel 4: WMMA cross-correlation.
// Grid: x = 16 row-blocks (16 u-rows each), y = dy (21), z = b (4).
// Block: 256 threads = 8 waves. Wave w handles dx = {3w,3w+1,3w+2} (24 padded).
// ---------------------------------------------------------------------------
__global__ __launch_bounds__(256, 1)
void k_corr(const half_t* __restrict__ cpy, float* __restrict__ out) {
    const int ub = blockIdx.x;
    const int dy = blockIdx.y;
    const int b  = blockIdx.z;
    const int wave = threadIdx.x >> 5;
    const int lane = threadIdx.x & 31;

    __shared__ half_t Brow[2][32 * BSTR];   // double-buffered, TDM-padded tiles

    v8f acc[9];
    for (int i = 0; i < 9; ++i)
        for (int e = 0; e < 8; ++e) acc[i][e] = 0.f;

    const int n_    = lane & 15;
    const int khalf = (lane >> 4) * 8;
    const int mhi   = (lane >> 4) * 8;

    const unsigned int lds0 = (unsigned int)(uintptr_t)(void*)&Brow[0][0];
    const unsigned int lds1 = (unsigned int)(uintptr_t)(void*)&Brow[1][0];

    // prologue: stage first B-row tile (r=0 copy, main region of row u)
    if (wave == 0) {
        tdm_stage(cpy + (((size_t)b * CC) * HH + ub * 16) * RS + 16, lds0);
    }

    for (int uu = 0; uu < 16; ++uu) {
        const int u = ub * 16 + uu;
        if (wave == 0) __builtin_amdgcn_s_wait_tensorcnt(0);  // buf[uu&1] landed
        __syncthreads();                                      // publish to all waves
        if (wave == 0 && uu + 1 < 16) {
            tdm_stage(cpy + (((size_t)b * CC) * HH + (u + 1)) * RS + 16,
                      ((uu + 1) & 1) ? lds1 : lds0);
        }

        const int ua = (u + dy - MS) & 255;    // dy-shifted A row (row-granular)

        const half_t* ap[3][2];
        #pragma unroll
        for (int di = 0; di < 3; ++di) {
            const int dx = wave * 3 + di;
            const int s  = dx - MS;            // column shift in [-10,13]
            const int r  = s & 3;
            const int qq = s - r;              // residual, multiple of 4
            #pragma unroll
            for (int it = 0; it < 2; ++it) {
                const int i_ = it * 16 + n_;
                ap[di][it] = cpy + ((((size_t)r * BB + b) * CC + i_) * HH + ua) * RS
                           + 16 + khalf + qq;
            }
        }
        const half_t* bp0 = Brow[uu & 1] + n_ * BSTR + khalf;
        const half_t* bp1 = bp0 + 16 * BSTR;

        #pragma unroll
        for (int kk = 0; kk < 8; ++kk) {
            const int kb = kk * 32;
            v16h Bf0 = combine8(*(const h8*)(bp0 + kb), *(const h8*)(bp0 + kb + 16));
            v16h Bf1 = combine8(*(const h8*)(bp1 + kb), *(const h8*)(bp1 + kb + 16));
            #pragma unroll
            for (int di = 0; di < 3; ++di) {
                v16h Af0 = loadA(ap[di][0], kb);
                v16h Af1 = loadA(ap[di][1], kb);
                acc[di * 3 + 0] = __builtin_amdgcn_wmma_f32_16x16x32_f16(
                    false, Af0, false, Bf0, (short)0, acc[di * 3 + 0], false, false);
                acc[di * 3 + 1] = __builtin_amdgcn_wmma_f32_16x16x32_f16(
                    false, Af0, false, Bf1, (short)0, acc[di * 3 + 1], false, false);
                acc[di * 3 + 2] = __builtin_amdgcn_wmma_f32_16x16x32_f16(
                    false, Af1, false, Bf1, (short)0, acc[di * 3 + 2], false, false);
            }
        }
    }

    // Writeback: upper-triangular pair packing; cross-block reduction via f32 atomics
    const int tI[3] = { 0, 0, 1 };
    const int tJ[3] = { 0, 1, 1 };
    for (int di = 0; di < 3; ++di) {
        const int dx = wave * 3 + di;
        if (dx >= NDX) continue;
        float* obase = out + ((((size_t)b * NDX + dy) * NDX) + dx) * NPAIR;
        for (int tt = 0; tt < 3; ++tt) {
            const int it = tI[tt], jt = tJ[tt];
            #pragma unroll
            for (int v = 0; v < 8; ++v) {
                const int i = it * 16 + mhi + v;
                const int j = jt * 16 + n_;
                if (i <= j) {
                    const int p = (i * (65 - i)) / 2 + (j - i);
                    atomicAdd(obase + p, acc[di * 3 + tt][v]);
                }
            }
        }
    }
}

// ---------------------------------------------------------------------------
// Launch
// ---------------------------------------------------------------------------
extern "C" void kernel_launch(void* const* d_in, const int* in_sizes, int n_in,
                              void* d_out, int out_size, void* d_ws, size_t ws_size,
                              hipStream_t stream) {
    (void)in_sizes; (void)n_in; (void)ws_size;
    const float* x = (const float*)d_in[0];
    float* out = (float*)d_out;
    float*  stats = (float*)d_ws;                       // 256 floats
    half_t* cpy   = (half_t*)((char*)d_ws + 1024);      // 4 halo'd copies, ~72 MiB

    k_stats <<<BB, 256, 0, stream>>>(x, stats);
    k_copies<<<dim3(HH, BB), 256, 0, stream>>>(x, stats, cpy);
    k_zero  <<<(out_size + 255) / 256, 256, 0, stream>>>(out, out_size);
    k_corr  <<<dim3(16, NDX, BB), 256, 0, stream>>>(cpy, out);
}
